// FGCNN_5970004542172
// MI455X (gfx1250) — compile-verified, matrix operand order checked by reading, not run on hardware
//
#include <hip/hip_runtime.h>
#include <hip/hip_bf16.h>
#include <math.h>

typedef __bf16 bf16_t;
typedef __attribute__((ext_vector_type(16))) __bf16 v16bf;
typedef __attribute__((ext_vector_type(8)))  float  v8f;

// Problem constants
#define BATCH   8192
#define NSPARSE 26
#define DDENSE  13
#define DIM     32
#define NF      8       // conv channels
#define TILE    16      // samples per workgroup (= WMMA M)
#define NTHREADS 128    // 4 waves

// Derived
#define K_FC0   3328    // 13*32*8
#define K_FC1   1536    // 6*32*8
#define K_MLP   1584
#define K_MLP_P 1600    // padded to multiple of 32
#define N_FC    96
#define N_MLP   128
#define AUGROWS 33
#define NPAIRS  528     // 33*32/2

// LDS layout (bytes)
#define OFF_X0    0                       // 16*3328 bf16  = 106496 B
#define OFF_EMB   106496                  // 16*27*32 f32  =  55296 B  (reused as s_hin: 16*1600 bf16 = 51200 B)
#define OFF_AUG   161792                  // 16*33*32 bf16 =  33792 B
#define OFF_X1    195584                  // 16*1536 bf16  =  49152 B
#define OFF_H     244736                  // 16*128  f32   =   8192 B
#define OFF_WT    252928                  // conv weights: 520 f32 = 2080 B
#define LDS_BYTES 255008

// ---- fast transcendental paths (hardware V_TANH_F32 / V_EXP_F32 on gfx1250)
__device__ __forceinline__ float fast_tanh(float x) {
#if __has_builtin(__builtin_amdgcn_tanhf)
    return __builtin_amdgcn_tanhf(x);
#elif __has_builtin(__builtin_amdgcn_tanh_f32)
    return __builtin_amdgcn_tanh_f32(x);
#else
    return tanhf(x);
#endif
}
__device__ __forceinline__ float fast_sigmoid(float x) {
#if __has_builtin(__builtin_amdgcn_exp2f)
    float e = __builtin_amdgcn_exp2f(-1.4426950408889634f * x);
#else
    float e = expf(-x);
#endif
    return 1.0f / (1.0f + e);
}
__device__ __forceinline__ float bf_lo(unsigned u) {
    union { unsigned i; float f; } c; c.i = u << 16; return c.f;
}
__device__ __forceinline__ float bf_hi(unsigned u) {
    union { unsigned i; float f; } c; c.i = u & 0xffff0000u; return c.f;
}

// ---------------------------------------------------------------------------
// Weight repack: fp32 (K x N) row-major -> bf16 tiles in WMMA B-operand lane
// order. Lane n = nt*16 + (lane&15); K of element e (pair p=e>>1):
//   k = kt*32 + ((p>>2)<<4) + (lane>>4)*8 + ((p&3)<<1) + (e&1)
// K beyond the true K is zero-filled (MLP pads 1584 -> 1600).
// ---------------------------------------------------------------------------
__global__ void pack_weights_bf16(const float* __restrict__ src, bf16_t* __restrict__ dst,
                                  int K, int N, int NTK, int NTN) {
    int total = NTK * NTN * 512;
    for (int idx = blockIdx.x * blockDim.x + threadIdx.x; idx < total;
         idx += gridDim.x * blockDim.x) {
        int e    = idx & 15;
        int lane = (idx >> 4) & 31;
        int tile = idx >> 9;
        int nt   = tile % NTN;
        int kt   = tile / NTN;
        int p    = e >> 1;
        int half = lane >> 4;
        int k = (kt << 5) + ((p >> 2) << 4) + (half << 3) + ((p & 3) << 1) + (e & 1);
        int n = (nt << 4) + (lane & 15);
        float v = (k < K && n < N) ? src[k * N + n] : 0.0f;
        dst[idx] = (bf16_t)v;
    }
}

// ---------------------------------------------------------------------------
// Per-wave GEMM: C(16x16) = A(16xKP bf16, row-major in LDS) * B(packed tiles)
// ---------------------------------------------------------------------------
__device__ __forceinline__ v8f wmma_gemm16(const bf16_t* sA, int KP,
                                           const bf16_t* __restrict__ packB,
                                           int nt, int NTN, int lane) {
    const int m    = lane & 15;
    const int half = lane >> 4;
    v8f c = {0.f, 0.f, 0.f, 0.f, 0.f, 0.f, 0.f, 0.f};
    const bf16_t* arow = sA + m * KP;
    const int nkt = KP >> 5;
    for (int kt = 0; kt < nkt; ++kt) {
        v16bf a, b;
        const uint4* pa = (const uint4*)(arow + (kt << 5) + (half << 3));
        ((uint4*)&a)[0] = pa[0];
        ((uint4*)&a)[1] = pa[2];   // +32 bytes = +16 bf16
        b = *(const v16bf*)(packB + ((((kt * NTN) + nt) << 5) + lane) * 16);
        c = __builtin_amdgcn_wmma_f32_16x16x32_bf16(false, a, false, b,
                                                    (short)0, c, false, false);
    }
    return c;
}

// ---------------------------------------------------------------------------
// Fully fused FGCNN forward: one workgroup = 16 samples, everything in LDS.
// ---------------------------------------------------------------------------
__global__ void __launch_bounds__(NTHREADS)
fgcnn_fused(const int*   __restrict__ sparse,   const float* __restrict__ dense,
            const float* __restrict__ gen_table, const float* __restrict__ gen_W,
            const float* __restrict__ gen_b,
            const float* __restrict__ cls_table, const float* __restrict__ cls_W,
            const float* __restrict__ cls_b,
            const float* __restrict__ ck0, const float* __restrict__ cb0,
            const float* __restrict__ fc_b0,
            const float* __restrict__ ck1, const float* __restrict__ cb1,
            const float* __restrict__ fc_b1,
            const float* __restrict__ mlp_b1,
            const float* __restrict__ bn_g, const float* __restrict__ bn_bt,
            const float* __restrict__ bn_m, const float* __restrict__ bn_v,
            const float* __restrict__ out_W, const float* __restrict__ out_b,
            const bf16_t* __restrict__ pW0, const bf16_t* __restrict__ pW1,
            const bf16_t* __restrict__ pW2,
            float* __restrict__ out) {
    extern __shared__ char smem[];
    bf16_t* s_x0  = (bf16_t*)(smem + OFF_X0);    // [16][13*32*8]
    float*  s_emb = (float*) (smem + OFF_EMB);   // [16][27][32]
    bf16_t* s_hin = (bf16_t*)(smem + OFF_EMB);   // [16][1600]  (reuses s_emb)
    bf16_t* s_aug = (bf16_t*)(smem + OFF_AUG);   // [16][33][32]
    bf16_t* s_x1  = (bf16_t*)(smem + OFF_X1);    // [16][6*32*8]
    float*  s_h   = (float*) (smem + OFF_H);     // [16][128]
    float*  s_wt  = (float*) (smem + OFF_WT);    // ck0(56)|cb0(8)|ck1(448)|cb1(8)

    const int tid   = threadIdx.x;
    const int lane  = tid & 31;
    const int wave  = tid >> 5;
    const int half  = lane >> 4;
    const int sBase = blockIdx.x * TILE;

    // ---- Phase 0: conv weights -> LDS -------------------------------------
    for (int i = tid; i < 520; i += NTHREADS) {
        float v;
        if      (i < 56)  v = ck0[i];
        else if (i < 64)  v = cb0[i - 56];
        else if (i < 512) v = ck1[i - 64];
        else              v = cb1[i - 512];
        s_wt[i] = v;
    }

    // ---- Phase 1a: gen embedding rows via ASYNC global->LDS (fp32) --------
    // 16 samples x 26 rows x (32 floats = 8 x 16B chunks) = 3328 chunks.
    for (int idx = tid; idx < TILE * NSPARSE * 8; idx += NTHREADS) {
        int q   = idx & 7;                 // 16B chunk within row
        int row = (idx >> 3) % NSPARSE;
        int m   = idx / (NSPARSE * 8);
        int tok = sparse[(sBase + m) * NSPARSE + row];
        unsigned long long gaddr =
            (unsigned long long)(uintptr_t)(gen_table + (size_t)tok * DIM + q * 4);
        unsigned int laddr =
            (unsigned int)(uintptr_t)(s_emb + (m * 27 + row) * DIM + q * 4);
        asm volatile("global_load_async_to_lds_b128 %0, %1, off"
                     :: "v"(laddr), "v"(gaddr) : "memory");
    }
    // dense row 26 (tiny 13->32 GEMV)
    for (int idx = tid; idx < TILE * DIM; idx += NTHREADS) {
        int d = idx & 31;
        int m = idx >> 5;
        float acc = gen_b[d];
        for (int k = 0; k < DDENSE; ++k)
            acc += dense[(sBase + m) * DDENSE + k] * gen_W[k * DIM + d];
        s_emb[(m * 27 + 26) * DIM + d] = acc;
    }

    // ---- Phase 1b: cls embedding -> aug rows 0..26 (bf16, float4 loads) ---
    for (int idx = tid; idx < TILE * NSPARSE * 8; idx += NTHREADS) {
        int q   = idx & 7;
        int row = (idx >> 3) % NSPARSE;
        int m   = idx / (NSPARSE * 8);
        int tok = sparse[(sBase + m) * NSPARSE + row];
        float4 v = *(const float4*)(cls_table + (size_t)tok * DIM + q * 4);
        bf16_t b4[4] = {(bf16_t)v.x, (bf16_t)v.y, (bf16_t)v.z, (bf16_t)v.w};
        *(uint2*)&s_aug[(m * AUGROWS + row) * DIM + q * 4] = *(const uint2*)b4;
    }
    for (int idx = tid; idx < TILE * DIM; idx += NTHREADS) {
        int d = idx & 31;
        int m = idx >> 5;
        float acc = cls_b[d];
        for (int k = 0; k < DDENSE; ++k)
            acc += dense[(sBase + m) * DDENSE + k] * cls_W[k * DIM + d];
        s_aug[(m * AUGROWS + 26) * DIM + d] = (bf16_t)acc;
    }
    asm volatile("s_wait_asynccnt 0" ::: "memory");
    __syncthreads();

    // ---- Phase 2: conv0 (7x1, 1->8) + tanh + maxpool(27->13) -> s_x0 ------
    for (int idx = tid; idx < TILE * K_FC0; idx += NTHREADS) {
        int f = idx & 7;
        int d = (idx >> 3) & 31;
        int p = (idx >> 8) % 13;
        int m = idx / K_FC0;
        const float* e = &s_emb[m * 27 * DIM + d];   // stride DIM over h
        float best = -1e30f;
        for (int hh = 2 * p; hh <= 2 * p + 1; ++hh) {
            float acc = s_wt[56 + f];
            #pragma unroll
            for (int t = 0; t < 7; ++t) {
                int h = hh + t - 3;
                if (h >= 0 && h < 27) acc += e[h * DIM] * s_wt[t * NF + f];
            }
            best = fmaxf(best, fast_tanh(acc));
        }
        s_x0[m * K_FC0 + (p * DIM + d) * NF + f] = (bf16_t)best;
    }
    __syncthreads();

    // ---- Phase 3: fc0 (16 x 3328 x 96) WMMA, tanh -> aug rows 27..29 ------
    for (int nt = wave; nt < N_FC / 16; nt += 4) {
        v8f c = wmma_gemm16(s_x0, K_FC0, pW0, nt, N_FC / 16, lane);
        int n = nt * 16 + (lane & 15);
        float bias = fc_b0[n];
        int row = 27 + (n >> 5);
        int d   = n & 31;
        #pragma unroll
        for (int r = 0; r < 8; ++r) {
            int mrow = r + 8 * half;
            s_aug[(mrow * AUGROWS + row) * DIM + d] = (bf16_t)fast_tanh(c[r] + bias);
        }
    }
    __syncthreads();

    // ---- Phase 4: conv1 (7x1, 8->8) + tanh + maxpool(13->6) -> s_x1 -------
    for (int idx = tid; idx < TILE * K_FC1; idx += NTHREADS) {
        int fo = idx & 7;
        int d  = (idx >> 3) & 31;
        int p  = (idx >> 8) % 6;
        int m  = idx / K_FC1;
        float best = -1e30f;
        for (int hh = 2 * p; hh <= 2 * p + 1; ++hh) {
            float acc = s_wt[512 + fo];
            #pragma unroll
            for (int t = 0; t < 7; ++t) {
                int h = hh + t - 3;
                if (h < 0 || h >= 13) continue;
                // 8 consecutive bf16 (16B aligned) -> one b128 LDS load
                uint4 xv = *(const uint4*)&s_x0[m * K_FC0 + (h * DIM + d) * NF];
                const float* kk = &s_wt[64 + t * 64 + fo];
                acc += bf_lo(xv.x) * kk[0]  + bf_hi(xv.x) * kk[8]
                     + bf_lo(xv.y) * kk[16] + bf_hi(xv.y) * kk[24]
                     + bf_lo(xv.z) * kk[32] + bf_hi(xv.z) * kk[40]
                     + bf_lo(xv.w) * kk[48] + bf_hi(xv.w) * kk[56];
            }
            best = fmaxf(best, fast_tanh(acc));
        }
        s_x1[m * K_FC1 + (p * DIM + d) * NF + fo] = (bf16_t)best;
    }
    __syncthreads();

    // ---- Phase 5: fc1 (16 x 1536 x 96) WMMA, tanh -> aug rows 30..32 ------
    for (int nt = wave; nt < N_FC / 16; nt += 4) {
        v8f c = wmma_gemm16(s_x1, K_FC1, pW1, nt, N_FC / 16, lane);
        int n = nt * 16 + (lane & 15);
        float bias = fc_b1[n];
        int row = 30 + (n >> 5);
        int d   = n & 31;
        #pragma unroll
        for (int r = 0; r < 8; ++r) {
            int mrow = r + 8 * half;
            s_aug[(mrow * AUGROWS + row) * DIM + d] = (bf16_t)fast_tanh(c[r] + bias);
        }
    }
    __syncthreads();

    // ---- Phase 6: Gram upper-triangle + aug flatten -> s_hin (bf16) -------
    for (int idx = tid; idx < TILE * NPAIRS; idx += NTHREADS) {
        int m = idx / NPAIRS;
        int p = idx % NPAIRS;
        int i = 0, rem = p;
        while (rem >= 32 - i) { rem -= 32 - i; ++i; }   // triu_indices(33, k=1)
        int j = i + 1 + rem;
        const uint4* ra = (const uint4*)&s_aug[(m * AUGROWS + i) * DIM];
        const uint4* rb = (const uint4*)&s_aug[(m * AUGROWS + j) * DIM];
        float acc = 0.0f;
        #pragma unroll
        for (int c = 0; c < 4; ++c) {
            uint4 va = ra[c], vb = rb[c];
            acc += bf_lo(va.x) * bf_lo(vb.x) + bf_hi(va.x) * bf_hi(vb.x)
                 + bf_lo(va.y) * bf_lo(vb.y) + bf_hi(va.y) * bf_hi(vb.y)
                 + bf_lo(va.z) * bf_lo(vb.z) + bf_hi(va.z) * bf_hi(vb.z)
                 + bf_lo(va.w) * bf_lo(vb.w) + bf_hi(va.w) * bf_hi(vb.w);
        }
        s_hin[m * K_MLP_P + p] = (bf16_t)acc;
    }
    // aug flatten + zero pad, 16B chunks (1056 bf16 = 132 chunks, + 2 pad chunks)
    for (int idx = tid; idx < TILE * 134; idx += NTHREADS) {
        int m = idx / 134;
        int ch = idx % 134;
        uint4 v;
        if (ch < 132) v = *(const uint4*)&s_aug[m * AUGROWS * DIM + ch * 8];
        else          v = make_uint4(0u, 0u, 0u, 0u);
        *(uint4*)&s_hin[m * K_MLP_P + NPAIRS + ch * 8] = v;
    }
    __syncthreads();

    // ---- Phase 7: MLP (16 x 1600 x 128) WMMA + relu + BN -> s_h -----------
    for (int nt = wave; nt < N_MLP / 16; nt += 4) {
        v8f c = wmma_gemm16(s_hin, K_MLP_P, pW2, nt, N_MLP / 16, lane);
        int j = nt * 16 + (lane & 15);
        float bias  = mlp_b1[j];
        float scale = bn_g[j] * rsqrtf(bn_v[j] + 0.001f);
        float mean  = bn_m[j];
        float beta  = bn_bt[j];
        #pragma unroll
        for (int r = 0; r < 8; ++r) {
            float v = fmaxf(c[r] + bias, 0.0f);
            s_h[(r + 8 * half) * N_MLP + j] = (v - mean) * scale + beta;
        }
    }
    __syncthreads();

    // ---- Phase 8: output head (128 -> 1) + sigmoid ------------------------
    if (tid < TILE) {
        float acc = out_b[0];
        #pragma unroll 8
        for (int j = 0; j < N_MLP; ++j) acc += s_h[tid * N_MLP + j] * out_W[j];
        out[sBase + tid] = fast_sigmoid(acc);
    }
}

// ---------------------------------------------------------------------------
extern "C" void kernel_launch(void* const* d_in, const int* in_sizes, int n_in,
                              void* d_out, int out_size, void* d_ws, size_t ws_size,
                              hipStream_t stream) {
    const int*   sparse    = (const int*)  d_in[0];
    const float* dense     = (const float*)d_in[1];
    const float* gen_table = (const float*)d_in[2];
    const float* gen_W     = (const float*)d_in[3];
    const float* gen_b     = (const float*)d_in[4];
    const float* cls_table = (const float*)d_in[5];
    const float* cls_W     = (const float*)d_in[6];
    const float* cls_b     = (const float*)d_in[7];
    const float* conv_k0   = (const float*)d_in[8];
    const float* conv_b0   = (const float*)d_in[9];
    const float* fc_W0     = (const float*)d_in[10];
    const float* fc_b0     = (const float*)d_in[11];
    const float* conv_k1   = (const float*)d_in[12];
    const float* conv_b1   = (const float*)d_in[13];
    const float* fc_W1     = (const float*)d_in[14];
    const float* fc_b1     = (const float*)d_in[15];
    const float* mlp_W1    = (const float*)d_in[16];
    const float* mlp_b1    = (const float*)d_in[17];
    const float* bn_gamma  = (const float*)d_in[18];
    const float* bn_beta   = (const float*)d_in[19];
    const float* bn_mean   = (const float*)d_in[20];
    const float* bn_var    = (const float*)d_in[21];
    const float* out_W     = (const float*)d_in[22];
    const float* out_b     = (const float*)d_in[23];
    float* out = (float*)d_out;

    // Packed bf16 weight tiles in workspace
    const size_t SZ_W0 = (size_t)(K_FC0 / 32) * (N_FC  / 16) * 512 * sizeof(bf16_t); // 638976
    const size_t SZ_W1 = (size_t)(K_FC1 / 32) * (N_FC  / 16) * 512 * sizeof(bf16_t); // 294912
    bf16_t* pW0 = (bf16_t*)d_ws;
    bf16_t* pW1 = (bf16_t*)((char*)d_ws + SZ_W0);
    bf16_t* pW2 = (bf16_t*)((char*)d_ws + SZ_W0 + SZ_W1);

    {
        int tot0 = (K_FC0 / 32) * (N_FC / 16) * 512;
        int tot1 = (K_FC1 / 32) * (N_FC / 16) * 512;
        int tot2 = (K_MLP_P / 32) * (N_MLP / 16) * 512;
        pack_weights_bf16<<<(tot0 + 255) / 256, 256, 0, stream>>>(fc_W0, pW0, K_FC0, N_FC, K_FC0 / 32, N_FC / 16);
        pack_weights_bf16<<<(tot1 + 255) / 256, 256, 0, stream>>>(fc_W1, pW1, K_FC1, N_FC, K_FC1 / 32, N_FC / 16);
        pack_weights_bf16<<<(tot2 + 255) / 256, 256, 0, stream>>>(mlp_W1, pW2, K_MLP, N_MLP, K_MLP_P / 32, N_MLP / 16);
    }

    fgcnn_fused<<<BATCH / TILE, NTHREADS, LDS_BYTES, stream>>>(
        sparse, dense, gen_table, gen_W, gen_b, cls_table, cls_W, cls_b,
        conv_k0, conv_b0, fc_b0, conv_k1, conv_b1, fc_b1,
        mlp_b1, bn_gamma, bn_beta, bn_mean, bn_var, out_W, out_b,
        pW0, pW1, pW2, out);
}
